// _BertCRF_86174223827057
// MI455X (gfx1250) — compile-verified, matrix operand order checked by reading, not run on hardware
//
#include <hip/hip_runtime.h>
#include <hip/hip_bf16.h>
#include <math.h>

// ---------------------------------------------------------------------------
// Model constants (BERT-base + CRF head)
// ---------------------------------------------------------------------------
#define LAYERS 12
#define H      768
#define NHEAD  12
#define DH     64          // H / NHEAD
#define TSEQ   512
#define BATCH  4
#define KTAGS  21
#define ROWS   (BATCH*TSEQ)   // 2048
#define NEGBIG (-10000.0f)
#define LN_EPS 1e-12f
#define INV_SQRT_DH 0.125f

// GEMM block tile: 128x64, 8 waves (4M x 2N), each wave a 32x32 register tile
// (2x2 WMMA accumulators) -> 4 v_wmma per wave per K-step of 32.
#define BM 128
#define BN 64
#define BK 32
#define AP (BK + 2)   // padded K-stride in LDS (bf16 units, even -> 4B aligned)

typedef __attribute__((ext_vector_type(16))) __bf16 v16bf;
typedef __attribute__((ext_vector_type(2)))  __bf16 v2bf;
typedef __attribute__((ext_vector_type(8)))  float  v8f;

// ---------------------------------------------------------------------------
// LDS-tiled GEMM: C = act(alpha * A@B + bias).
// A: MxK fp32 row-major (lda), 16B-aligned rows. B: KxN fp32; transB=1 means
// B[k][n]=Bp[n*ldb+k]. K multiple of BK (768/3072/64/512). M multiple of BM.
// Software-pipelined: K-tile (t+1) is fetched global->regs while WMMAs consume
// K-tile (t) from LDS. Loads are unconditional from clamped addresses with
// value-selects (v_cndmask), never predicated loads.
// Fragment layouts per CDNA5 ISA 7.12.2 (wave32):
//   16-bit A 16x32 : lane m=lane&15; lanes16-31 carry K+8 / K+24 halves.
//   C/D 16x16 f32  : VGPR r -> M = r + 8*(lane>>4), N = lane&15.
// ---------------------------------------------------------------------------
__device__ __forceinline__ void block_tile_gemm(
    const float* __restrict__ A, int lda,
    const float* __restrict__ B, int ldb, int transB,
    float* __restrict__ C, int ldc,
    int M, int N, int Kd,
    int rowB, int colB,              // block-tile origin
    float alpha, const float* __restrict__ bias, int act)
{
    __shared__ __bf16 As[BM][AP];    // [m][k]
    __shared__ __bf16 Bt[BN][AP];    // [n][k]  (B transposed tile)

    const int tid  = threadIdx.x;
    const int lane = tid & 31;
    const int wave = tid >> 5;       // 0..7
    const int wm   = wave >> 1;      // 0..3  (M)
    const int wn   = wave & 1;       // 0..1  (N)
    const int half = lane >> 4;      // K-half select
    const int mloc = lane & 15;

    // A staging: 2 threads per row, 16 consecutive K floats each (4 x float4)
    const int a_r  = tid >> 1;             // 0..127
    const int a_kq = (tid & 1) * 16;       // 0 or 16
    const int ar   = rowB + a_r;
    const int arc  = (ar < M) ? ar : (M - 1);
    const float a_ok = (ar < M) ? 1.0f : 0.0f;

    // B staging coords
    const int bt_n  = tid >> 2;            // transB: 0..63
    const int bt_kq = (tid & 3) * 8;       // transB: 0,8,16,24
    const int bn_kx = tid >> 3;            // normal: 0..31
    const int bn_n8 = (tid & 7) * 8;       // normal: 0,8,..,56

    float aR[16];
    float bR[8];

    auto loadA = [&](int kk) {
        const float4* ap = reinterpret_cast<const float4*>(
            A + (size_t)arc * lda + kk + a_kq);
#pragma unroll
        for (int q = 0; q < 4; ++q) {
            float4 f = ap[q];
            aR[4 * q + 0] = f.x * a_ok;
            aR[4 * q + 1] = f.y * a_ok;
            aR[4 * q + 2] = f.z * a_ok;
            aR[4 * q + 3] = f.w * a_ok;
        }
    };
    auto loadB = [&](int kk) {
        if (transB) {
            const int bn  = colB + bt_n;
            const int bnc = (bn < N) ? bn : (N - 1);
            const float ok = (bn < N) ? 1.0f : 0.0f;
            const float4* bp = reinterpret_cast<const float4*>(
                B + (size_t)bnc * ldb + kk + bt_kq);
#pragma unroll
            for (int q = 0; q < 2; ++q) {
                float4 f = bp[q];
                bR[4 * q + 0] = f.x * ok;
                bR[4 * q + 1] = f.y * ok;
                bR[4 * q + 2] = f.z * ok;
                bR[4 * q + 3] = f.w * ok;
            }
        } else {
            const float* bp = B + (size_t)(kk + bn_kx) * ldb;
#pragma unroll
            for (int i = 0; i < 8; ++i) {
                const int bn  = colB + bn_n8 + i;
                const int bnc = (bn < N) ? bn : (N - 1);
                float f = bp[bnc];
                bR[i] = (bn < N) ? f : 0.0f;
            }
        }
    };
    auto storeA = [&]() {
#pragma unroll
        for (int q = 0; q < 8; ++q) {
            v2bf p; p.x = (__bf16)aR[2 * q]; p.y = (__bf16)aR[2 * q + 1];
            *reinterpret_cast<v2bf*>(&As[a_r][a_kq + 2 * q]) = p;
        }
    };
    auto storeB = [&]() {
        if (transB) {
#pragma unroll
            for (int q = 0; q < 4; ++q) {
                v2bf p; p.x = (__bf16)bR[2 * q]; p.y = (__bf16)bR[2 * q + 1];
                *reinterpret_cast<v2bf*>(&Bt[bt_n][bt_kq + 2 * q]) = p;
            }
        } else {
#pragma unroll
            for (int i = 0; i < 8; ++i)
                Bt[bn_n8 + i][bn_kx] = (__bf16)bR[i];
        }
    };

    v8f acc[2][2] = {};

    loadA(0);
    loadB(0);
    for (int kk = 0; kk < Kd; kk += BK) {
        storeA();
        storeB();
        __syncthreads();

        // prefetch next K-tile into registers while WMMAs run on this one
        if (kk + BK < Kd) { loadA(kk + BK); loadB(kk + BK); }

        const int ar0 = wm * 32 + mloc;
        const int br0 = wn * 32 + mloc;
        v16bf a0, a1, b0, b1;
#pragma unroll
        for (int i = 0; i < 8; ++i) {
            const int kof = ((i < 4) ? (2 * i) : (16 + 2 * (i - 4))) + half * 8;
            v2bf pa0 = *reinterpret_cast<const v2bf*>(&As[ar0][kof]);
            v2bf pa1 = *reinterpret_cast<const v2bf*>(&As[ar0 + 16][kof]);
            v2bf pb0 = *reinterpret_cast<const v2bf*>(&Bt[br0][kof]);
            v2bf pb1 = *reinterpret_cast<const v2bf*>(&Bt[br0 + 16][kof]);
            a0[2 * i] = pa0.x; a0[2 * i + 1] = pa0.y;
            a1[2 * i] = pa1.x; a1[2 * i + 1] = pa1.y;
            b0[2 * i] = pb0.x; b0[2 * i + 1] = pb0.y;
            b1[2 * i] = pb1.x; b1[2 * i + 1] = pb1.y;
        }
        acc[0][0] = __builtin_amdgcn_wmma_f32_16x16x32_bf16(
            false, a0, false, b0, (short)0, acc[0][0], false, false);
        acc[0][1] = __builtin_amdgcn_wmma_f32_16x16x32_bf16(
            false, a0, false, b1, (short)0, acc[0][1], false, false);
        acc[1][0] = __builtin_amdgcn_wmma_f32_16x16x32_bf16(
            false, a1, false, b0, (short)0, acc[1][0], false, false);
        acc[1][1] = __builtin_amdgcn_wmma_f32_16x16x32_bf16(
            false, a1, false, b1, (short)0, acc[1][1], false, false);
        __syncthreads();
    }

    // ---- epilogue ----
#pragma unroll
    for (int mi = 0; mi < 2; ++mi) {
#pragma unroll
        for (int ni = 0; ni < 2; ++ni) {
            const int row0 = rowB + wm * 32 + mi * 16;
            const int col0 = colB + wn * 32 + ni * 16;
#pragma unroll
            for (int r = 0; r < 8; ++r) {
                const int m = row0 + r + 8 * half;
                const int n = col0 + mloc;
                if (m < M && n < N) {
                    float v = acc[mi][ni][r] * alpha;
                    if (bias) v += bias[n];
                    if (act == 1)  // exact GELU: 0.5*x*(1+erf(x/sqrt(2)))
                        v = 0.5f * v * (1.0f + erff(v * 0.70710678118654752f));
                    C[(size_t)m * ldc + n] = v;
                }
            }
        }
    }
}

__global__ void gemm_kernel(const float* __restrict__ A, int lda,
                            const float* __restrict__ B, int ldb, int transB,
                            float* __restrict__ C, int ldc,
                            int M, int N, int Kd, float alpha,
                            const float* __restrict__ bias, int act)
{
    block_tile_gemm(A, lda, B, ldb, transB, C, ldc, M, N, Kd,
                    blockIdx.y * BM, blockIdx.x * BN, alpha, bias, act);
}

// scores[bh, q, k] = (1/sqrt(DH)) * Q[bh] @ K[bh]^T      (grid.z = b*NHEAD + h)
__global__ void attn_scores_kernel(const float* __restrict__ qkv,
                                   float* __restrict__ scores)
{
    const int bh = blockIdx.z;
    const int b  = bh / NHEAD;
    const int hh = bh % NHEAD;
    const float* Abase = qkv + (size_t)b * TSEQ * 3 * H + hh * DH;         // Q
    const float* Bbase = qkv + (size_t)b * TSEQ * 3 * H + H + hh * DH;     // K (transB)
    float*       Cbase = scores + (size_t)bh * TSEQ * TSEQ;
    block_tile_gemm(Abase, 3 * H, Bbase, 3 * H, 1, Cbase, TSEQ,
                    TSEQ, TSEQ, DH, blockIdx.y * BM, blockIdx.x * BN,
                    INV_SQRT_DH, nullptr, 0);
}

// ctx[b, t, h*DH + d] = P[bh] @ V[bh]   (written directly in [B,T,H] layout)
__global__ void attn_ctx_kernel(const float* __restrict__ probs,
                                const float* __restrict__ qkv,
                                float* __restrict__ ctx)
{
    const int bh = blockIdx.z;
    const int b  = bh / NHEAD;
    const int hh = bh % NHEAD;
    const float* Abase = probs + (size_t)bh * TSEQ * TSEQ;
    const float* Bbase = qkv + (size_t)b * TSEQ * 3 * H + 2 * H + hh * DH;  // V
    float*       Cbase = ctx + (size_t)b * TSEQ * H + hh * DH;
    block_tile_gemm(Abase, TSEQ, Bbase, 3 * H, 0, Cbase, H,
                    TSEQ, DH, TSEQ, blockIdx.y * BM, blockIdx.x * BN,
                    1.0f, nullptr, 0);
}

// Row softmax with additive attention-mask bias. grid = (TSEQ, B*NHEAD), 256 thr.
__global__ void softmax_kernel(float* __restrict__ scores,
                               const int* __restrict__ attn_mask)
{
    const int q  = blockIdx.x;
    const int bh = blockIdx.y;
    const int b  = bh / NHEAD;
    float* row = scores + ((size_t)bh * TSEQ + q) * TSEQ;
    const int* mrow = attn_mask + b * TSEQ;
    __shared__ float sm[256];
    const int tid = threadIdx.x;

    float vals[2];
    float mx = -INFINITY;
#pragma unroll
    for (int i = 0; i < 2; ++i) {
        const int k = tid + i * 256;
        float v = row[k] + (1.0f - (float)mrow[k]) * NEGBIG;
        vals[i] = v;
        mx = fmaxf(mx, v);
    }
    sm[tid] = mx; __syncthreads();
    for (int s = 128; s > 0; s >>= 1) { if (tid < s) sm[tid] = fmaxf(sm[tid], sm[tid + s]); __syncthreads(); }
    mx = sm[0]; __syncthreads();

    float sum = 0.0f;
#pragma unroll
    for (int i = 0; i < 2; ++i) { vals[i] = expf(vals[i] - mx); sum += vals[i]; }
    sm[tid] = sum; __syncthreads();
    for (int s = 128; s > 0; s >>= 1) { if (tid < s) sm[tid] += sm[tid + s]; __syncthreads(); }
    const float inv = 1.0f / sm[0];
#pragma unroll
    for (int i = 0; i < 2; ++i) row[tid + i * 256] = vals[i] * inv;
}

// h = LN(h + x) * g + b   (x may be null -> LN(h)). One block per row, 256 thr.
__global__ void add_ln_kernel(float* __restrict__ h,
                              const float* __restrict__ x,
                              const float* __restrict__ g,
                              const float* __restrict__ bta)
{
    const int row = blockIdx.x;
    float* hr = h + (size_t)row * H;
    const float* xr = x ? (x + (size_t)row * H) : nullptr;
    __shared__ float sm[256];
    const int tid = threadIdx.x;

    float y[3];
    float s = 0.0f, ss = 0.0f;
#pragma unroll
    for (int i = 0; i < 3; ++i) {
        const int idx = tid + i * 256;
        float v = hr[idx] + (xr ? xr[idx] : 0.0f);
        y[i] = v; s += v; ss += v * v;
    }
    sm[tid] = s; __syncthreads();
    for (int t = 128; t > 0; t >>= 1) { if (tid < t) sm[tid] += sm[tid + t]; __syncthreads(); }
    const float mean = sm[0] * (1.0f / H); __syncthreads();
    sm[tid] = ss; __syncthreads();
    for (int t = 128; t > 0; t >>= 1) { if (tid < t) sm[tid] += sm[tid + t]; __syncthreads(); }
    const float var = sm[0] * (1.0f / H) - mean * mean;
    const float inv = rsqrtf(var + LN_EPS);
#pragma unroll
    for (int i = 0; i < 3; ++i) {
        const int idx = tid + i * 256;
        hr[idx] = (y[i] - mean) * inv * g[idx] + bta[idx];
    }
}

// h0 = word_emb[id] + pos_emb[t] + tok_emb[0]
__global__ void embed_kernel(const int* __restrict__ ids,
                             const float* __restrict__ we,
                             const float* __restrict__ pe,
                             const float* __restrict__ te,
                             float* __restrict__ h)
{
    const int row = blockIdx.x;
    const int t   = row % TSEQ;
    const int id  = ids[row];
    const int tid = threadIdx.x;
#pragma unroll
    for (int i = 0; i < 3; ++i) {
        const int idx = tid + i * 256;
        h[(size_t)row * H + idx] =
            we[(size_t)id * H + idx] + pe[(size_t)t * H + idx] + te[idx];
    }
}

// ---------------------------------------------------------------------------
// CRF: NLL (forward logsumexp) + Viterbi decode. One wave; trivially small.
// out[0] = nll, out[1 + b*T + t] = (float)decoded tag.
// ---------------------------------------------------------------------------
__global__ void crf_kernel(const float* __restrict__ em,      // [B,T,K]
                           const int* __restrict__ mask,      // [B,T]
                           const int* __restrict__ labels,    // [B,T]
                           const float* __restrict__ start,
                           const float* __restrict__ endv,
                           const float* __restrict__ trans,   // [K,K]
                           int* __restrict__ hist,            // [B,T,K] ws
                           float* __restrict__ out)
{
    const int tid = threadIdx.x;
    __shared__ float alpha[KTAGS], nxt[KTAGS], score[KTAGS], snxt[KTAGS];
    __shared__ float nll_acc;
    if (tid == 0) nll_acc = 0.0f;
    __syncthreads();

    for (int b = 0; b < BATCH; ++b) {
        const float* emB = em + (size_t)b * TSEQ * KTAGS;
        const int*   mB  = mask + b * TSEQ;
        const int*   lB  = labels + b * TSEQ;

        // ---- forward algorithm (denominator) ----
        if (tid < KTAGS) alpha[tid] = start[tid] + emB[tid];
        __syncthreads();
        for (int t = 1; t < TSEQ; ++t) {
            if (tid < KTAGS) {
                float mx = -INFINITY;
                for (int i = 0; i < KTAGS; ++i)
                    mx = fmaxf(mx, alpha[i] + trans[i * KTAGS + tid]);
                float s = 0.0f;
                for (int i = 0; i < KTAGS; ++i)
                    s += expf(alpha[i] + trans[i * KTAGS + tid] - mx);
                const float v = mx + logf(s) + emB[(size_t)t * KTAGS + tid];
                nxt[tid] = mB[t] ? v : alpha[tid];
            }
            __syncthreads();
            if (tid < KTAGS) alpha[tid] = nxt[tid];
            __syncthreads();
        }
        if (tid == 0) {
            float mx = -INFINITY;
            for (int j = 0; j < KTAGS; ++j) mx = fmaxf(mx, alpha[j] + endv[j]);
            float s = 0.0f;
            for (int j = 0; j < KTAGS; ++j) s += expf(alpha[j] + endv[j] - mx);
            const float den = mx + logf(s);
            // numerator
            float num = start[lB[0]] + emB[lB[0]];
            int cnt = 0;
            for (int t = 0; t < TSEQ; ++t) cnt += mB[t];
            for (int t = 1; t < TSEQ; ++t)
                if (mB[t])
                    num += trans[lB[t - 1] * KTAGS + lB[t]] +
                           emB[(size_t)t * KTAGS + lB[t]];
            num += endv[lB[cnt - 1]];
            nll_acc += -(num - den);
        }
        __syncthreads();

        // ---- Viterbi ----
        if (tid < KTAGS) score[tid] = start[tid] + emB[tid];
        __syncthreads();
        for (int t = 1; t < TSEQ; ++t) {
            if (tid < KTAGS) {
                float bv = -INFINITY; int bidx = 0;
                for (int i = 0; i < KTAGS; ++i) {
                    const float c = score[i] + trans[i * KTAGS + tid];
                    if (c > bv) { bv = c; bidx = i; }   // first-max like argmax
                }
                if (mB[t]) {
                    snxt[tid] = bv + emB[(size_t)t * KTAGS + tid];
                    hist[((size_t)b * TSEQ + t) * KTAGS + tid] = bidx;
                } else {
                    snxt[tid] = score[tid];
                    hist[((size_t)b * TSEQ + t) * KTAGS + tid] = tid;
                }
            }
            __syncthreads();
            if (tid < KTAGS) score[tid] = snxt[tid];
            __syncthreads();
        }
        if (tid == 0) {
            float bv = -INFINITY; int tag = 0;
            for (int j = 0; j < KTAGS; ++j) {
                const float c = score[j] + endv[j];
                if (c > bv) { bv = c; tag = j; }
            }
            out[1 + b * TSEQ + (TSEQ - 1)] = (float)tag;
            for (int t = TSEQ - 1; t >= 1; --t) {
                tag = hist[((size_t)b * TSEQ + t) * KTAGS + tag];
                out[1 + b * TSEQ + t - 1] = (float)tag;
            }
        }
        __syncthreads();
    }
    if (tid == 0) out[0] = nll_acc / (float)BATCH;
}

// ---------------------------------------------------------------------------
// Host launcher
// ---------------------------------------------------------------------------
static inline dim3 gemm_grid(int M, int N) {
    return dim3((N + BN - 1) / BN, (M + BM - 1) / BM, 1);
}

extern "C" void kernel_launch(void* const* d_in, const int* in_sizes, int n_in,
                              void* d_out, int out_size, void* d_ws, size_t ws_size,
                              hipStream_t stream) {
    (void)in_sizes; (void)n_in; (void)out_size; (void)ws_size;
    const int*   input_ids = (const int*)  d_in[0];
    const int*   attn_mask = (const int*)  d_in[1];
    const int*   labels    = (const int*)  d_in[3];
    const float* word_emb  = (const float*)d_in[4];
    const float* pos_emb   = (const float*)d_in[5];
    const float* tok_emb   = (const float*)d_in[6];
    const float* emb_g     = (const float*)d_in[7];
    const float* emb_b     = (const float*)d_in[8];
    const float* Wqkv      = (const float*)d_in[9];
    const float* bqkv      = (const float*)d_in[10];
    const float* Wo        = (const float*)d_in[11];
    const float* bo        = (const float*)d_in[12];
    const float* ln1_g     = (const float*)d_in[13];
    const float* ln1_b     = (const float*)d_in[14];
    const float* W1        = (const float*)d_in[15];
    const float* b1        = (const float*)d_in[16];
    const float* W2        = (const float*)d_in[17];
    const float* b2        = (const float*)d_in[18];
    const float* ln2_g     = (const float*)d_in[19];
    const float* ln2_b     = (const float*)d_in[20];
    const float* Wc        = (const float*)d_in[21];
    const float* bc        = (const float*)d_in[22];
    const float* crf_start = (const float*)d_in[23];
    const float* crf_end   = (const float*)d_in[24];
    const float* crf_trans = (const float*)d_in[25];

    // Workspace layout (~107 MB of fp32)
    float* ws = (float*)d_ws;
    size_t off = 0;
    float* h      = ws + off; off += (size_t)ROWS * H;                     // 2048x768
    float* qkv    = ws + off; off += (size_t)ROWS * 3 * H;                 // 2048x2304
    float* scores = ws + off; off += (size_t)BATCH * NHEAD * TSEQ * TSEQ;  // 48x512x512
    float* ctx    = ws + off; off += (size_t)ROWS * H;                     // 2048x768
    float* tmp    = ws + off; off += (size_t)ROWS * 4 * H;                 // 2048x3072
    float* emis   = ws + off; off += (size_t)ROWS * KTAGS;                 // 2048x21
    int*   hist   = (int*)(ws + off);                                      // 4x512x21

    // Embedding + LN
    embed_kernel<<<ROWS, 256, 0, stream>>>(input_ids, word_emb, pos_emb, tok_emb, h);
    add_ln_kernel<<<ROWS, 256, 0, stream>>>(h, nullptr, emb_g, emb_b);

    for (int l = 0; l < LAYERS; ++l) {
        const float* Wqkv_l = Wqkv + (size_t)l * H * 3 * H;
        const float* bqkv_l = bqkv + (size_t)l * 3 * H;
        const float* Wo_l   = Wo   + (size_t)l * H * H;
        const float* bo_l   = bo   + (size_t)l * H;
        const float* g1_l   = ln1_g + (size_t)l * H;
        const float* b1g_l  = ln1_b + (size_t)l * H;
        const float* W1_l   = W1   + (size_t)l * H * 4 * H;
        const float* b1_l   = b1   + (size_t)l * 4 * H;
        const float* W2_l   = W2   + (size_t)l * 4 * H * H;
        const float* b2_l   = b2   + (size_t)l * H;
        const float* g2_l   = ln2_g + (size_t)l * H;
        const float* b2g_l  = ln2_b + (size_t)l * H;

        // qkv = h @ Wqkv + bqkv
        gemm_kernel<<<gemm_grid(ROWS, 3 * H), 256, 0, stream>>>(
            h, H, Wqkv_l, 3 * H, 0, qkv, 3 * H, ROWS, 3 * H, H, 1.0f, bqkv_l, 0);

        // scores = (Q @ K^T) / sqrt(DH)
        attn_scores_kernel<<<dim3(TSEQ / BN, TSEQ / BM, BATCH * NHEAD), 256, 0, stream>>>(qkv, scores);

        // softmax with mask bias
        softmax_kernel<<<dim3(TSEQ, BATCH * NHEAD), 256, 0, stream>>>(scores, attn_mask);

        // ctx = P @ V
        attn_ctx_kernel<<<dim3(1, TSEQ / BM, BATCH * NHEAD), 256, 0, stream>>>(scores, qkv, ctx);

        // tmp = ctx @ Wo + bo ; h = LN(h + tmp)
        gemm_kernel<<<gemm_grid(ROWS, H), 256, 0, stream>>>(
            ctx, H, Wo_l, H, 0, tmp, H, ROWS, H, H, 1.0f, bo_l, 0);
        add_ln_kernel<<<ROWS, 256, 0, stream>>>(h, tmp, g1_l, b1g_l);

        // tmp = gelu(h @ W1 + b1)
        gemm_kernel<<<gemm_grid(ROWS, 4 * H), 256, 0, stream>>>(
            h, H, W1_l, 4 * H, 0, tmp, 4 * H, ROWS, 4 * H, H, 1.0f, b1_l, 1);

        // ctx = tmp @ W2 + b2 ; h = LN(h + ctx)
        gemm_kernel<<<gemm_grid(ROWS, H), 256, 0, stream>>>(
            tmp, 4 * H, W2_l, H, 0, ctx, H, ROWS, H, 4 * H, 1.0f, b2_l, 0);
        add_ln_kernel<<<ROWS, 256, 0, stream>>>(h, ctx, g2_l, b2g_l);
    }

    // emissions = h @ Wc + bc
    gemm_kernel<<<gemm_grid(ROWS, KTAGS), 256, 0, stream>>>(
        h, H, Wc, KTAGS, 0, emis, KTAGS, ROWS, KTAGS, H, 1.0f, bc, 0);

    // CRF NLL + Viterbi
    crf_kernel<<<1, 32, 0, stream>>>(emis, attn_mask, labels,
                                     crf_start, crf_end, crf_trans,
                                     hist, (float*)d_out);
}
